// AutoInt_61332132987353
// MI455X (gfx1250) — compile-verified, matrix operand order checked by reading, not run on hardware
//
#include <hip/hip_runtime.h>
#include <math.h>

#define BATCH  8192
#define EMB    64
#define HD     128
#define HEADS  4
#define ATT    32
#define FIELDS 32

typedef __attribute__((ext_vector_type(16))) _Float16 v16h;
typedef __attribute__((ext_vector_type(8)))  _Float16 v8h;
typedef __attribute__((ext_vector_type(8)))  float    v8f;

// ---------------------------------------------------------------------------
// Fragment helpers (CDNA5 WMMA 16x16x32 f16 layouts, CDNA5 ISA)
// A 16x32 (MxK): lane<16 -> M=lane, halves 0..7 = K0..7, 8..15 = K16..23
//                lane>=16 -> M=lane-16, halves 0..7 = K8..15, 8..15 = K24..31
// C  16x16 f32 : VGPR r -> M = r + (lane<16 ? 0 : 8), N = lane&15
// ---------------------------------------------------------------------------
__device__ __forceinline__ v16h load_frag_rm(const _Float16* base, int ld,
                                             int rt, int kt, int lane) {
  int r  = rt * 16 + (lane & 15);
  int hi = (lane >> 4) & 1;
  const _Float16* p = base + r * ld + kt * 32 + hi * 8;
  v8h a = *(const v8h*)p;           // K 0..7   (+8 for upper lanes)
  v8h b = *(const v8h*)(p + 16);    // K 16..23 (+8 for upper lanes)
  v16h o;
#pragma unroll
  for (int i = 0; i < 8; ++i) { o[i] = a[i]; o[i + 8] = b[i]; }
  return o;
}

__device__ __forceinline__ v8f wmma_f16(v16h a, v16h b, v8f c) {
  return __builtin_amdgcn_wmma_f32_16x16x32_f16(false, a, false, b,
                                                (short)0, c, false, false);
}

// ---------------------------------------------------------------------------
// Weight pre-pack: W [N][K] f32 row-major -> f16 B-fragments.
// Fragment fi = nt*(K/32)+kt holds 32 lanes * 16 halves contiguously:
//   value(lane,h) = W[nt*16 + (lane&15)][kt*32 + klocal(lane,h)]
// ---------------------------------------------------------------------------
__global__ void pack_weights(const float* __restrict__ W,
                             _Float16* __restrict__ dst, int N, int K) {
  int t = blockIdx.x * blockDim.x + threadIdx.x;
  if (t >= N * K) return;
  int h    = t & 15;
  int lane = (t >> 4) & 31;
  int fi   = t >> 9;
  int ktiles = K >> 5;
  int nt = fi / ktiles, kt = fi % ktiles;
  int n = nt * 16 + (lane & 15);
  int k = kt * 32 + (h < 8 ? h : h + 8) + ((lane >> 4) & 1) * 8;
  dst[t] = (_Float16)W[n * K + k];
}

// ---------------------------------------------------------------------------
// Projection: dst = src(32xK) @ W^T + bias, via packed B fragments.
// MODE 0: row-major f16 (ld=HD)   1: transposed f16 (Vt[d][f])   2: f32 rm
// ---------------------------------------------------------------------------
template <int MODE>
__device__ __forceinline__ void project(const _Float16* src, int K,
                                        const _Float16* wfrag,
                                        const float* bias,
                                        _Float16* dstH, float* dstF,
                                        int lane) {
  const int ktiles = K >> 5;
  for (int nt = 0; nt < HD / 16; ++nt) {
    float bv = bias[nt * 16 + (lane & 15)];
    for (int mt = 0; mt < FIELDS / 16; ++mt) {
      v8f c;
#pragma unroll
      for (int i = 0; i < 8; ++i) c[i] = bv;
      for (int kt = 0; kt < ktiles; ++kt) {
        v16h a  = load_frag_rm(src, K, mt, kt, lane);
        v16h bf = *(const v16h*)(wfrag + ((nt * ktiles + kt) * 32 + lane) * 16);
        c = wmma_f16(a, bf, c);
      }
      int n     = nt * 16 + (lane & 15);
      int mbase = mt * 16 + ((lane >> 4) & 1) * 8;
#pragma unroll
      for (int r = 0; r < 8; ++r) {
        int m = mbase + r;
        if (MODE == 0)      dstH[m * HD + n]     = (_Float16)c[r];
        else if (MODE == 1) dstH[n * FIELDS + m] = (_Float16)c[r];
        else                dstF[m * HD + n]     = c[r];
      }
    }
  }
}

// ---------------------------------------------------------------------------
// Attention: softmax(Q Kt) V + Res, relu.  FINAL: fold into logit partial.
// ---------------------------------------------------------------------------
template <bool FINAL>
__device__ __forceinline__ float attn(const _Float16* Qm, const _Float16* Km,
                                      const _Float16* Vt, const float* Res,
                                      float* S, _Float16* Am,
                                      _Float16* y2out,
                                      const float* __restrict__ logitW,
                                      int lane) {
  float part = 0.f;
  for (int h = 0; h < HEADS; ++h) {
    // ---- scores S = Q_h (32xATT) @ K_h^T ------------------------------
#pragma unroll
    for (int mt = 0; mt < 2; ++mt)
#pragma unroll
      for (int nt = 0; nt < 2; ++nt) {
        v16h a  = load_frag_rm(Qm + h * ATT, HD, mt, 0, lane);
        v16h bf = load_frag_rm(Km + h * ATT, HD, nt, 0, lane);
        v8f c;
#pragma unroll
        for (int i = 0; i < 8; ++i) c[i] = 0.f;
        c = wmma_f16(a, bf, c);
        int n     = nt * 16 + (lane & 15);
        int mbase = mt * 16 + ((lane >> 4) & 1) * 8;
#pragma unroll
        for (int r = 0; r < 8; ++r) S[(mbase + r) * FIELDS + n] = c[r];
      }
    __syncthreads();
    // ---- softmax: one row per lane ------------------------------------
    {
      const float* row = S + lane * FIELDS;
      float mx = row[0];
#pragma unroll
      for (int j = 1; j < FIELDS; ++j) mx = fmaxf(mx, row[j]);
      float sum = 0.f;
#pragma unroll
      for (int j = 0; j < FIELDS; ++j) sum += __expf(row[j] - mx);
      float inv = 1.f / sum;
#pragma unroll
      for (int j = 0; j < FIELDS; ++j)
        Am[lane * FIELDS + j] = (_Float16)(__expf(row[j] - mx) * inv);
    }
    __syncthreads();
    // ---- O = A @ V_h, C preloaded with Res, relu ----------------------
#pragma unroll
    for (int mt = 0; mt < 2; ++mt)
#pragma unroll
      for (int nt = 0; nt < 2; ++nt) {
        int n     = nt * 16 + (lane & 15);
        int mbase = mt * 16 + ((lane >> 4) & 1) * 8;
        v8f c;
#pragma unroll
        for (int r = 0; r < 8; ++r)
          c[r] = Res[(mbase + r) * HD + h * ATT + n];
        v16h a  = load_frag_rm(Am, FIELDS, mt, 0, lane);
        v16h bf = load_frag_rm(Vt + h * ATT * FIELDS, FIELDS, nt, 0, lane);
        c = wmma_f16(a, bf, c);
#pragma unroll
        for (int r = 0; r < 8; ++r) {
          float v = fmaxf(c[r], 0.f);
          int m = mbase + r, ng = h * ATT + n;
          if (FINAL) part += v * logitW[m * HD + ng];
          else       y2out[m * HD + ng] = (_Float16)v;
        }
      }
    __syncthreads();   // S / Am recycled next head
  }
  return part;
}

// ---------------------------------------------------------------------------
// Fused AutoInt kernel: one wave32 per sample.
// ---------------------------------------------------------------------------
__global__ void __launch_bounds__(32) autoint_fused(
    const int* __restrict__ onehot_i, const float* __restrict__ onehot_x,
    const int* __restrict__ mh_i, const float* __restrict__ mh_x,
    const float* __restrict__ ctns, const float* __restrict__ xx,
    const float* __restrict__ xy,
    const float* Qb1, const float* Kb1, const float* Vb1, const float* Rb1,
    const float* Qb2, const float* Kb2, const float* Vb2, const float* Rb2,
    const float* __restrict__ logitW, const float* __restrict__ logitb,
    const _Float16* __restrict__ wpack, float* __restrict__ out) {
  __shared__ _Float16 yh[FIELDS * EMB];    // 4 KB  layer-1 input, f16
  __shared__ _Float16 Qm[FIELDS * HD];     // 8 KB
  __shared__ _Float16 Km[FIELDS * HD];     // 8 KB
  __shared__ _Float16 Vt[HD * FIELDS];     // 8 KB  V transposed [d][f]
  __shared__ float    Res[FIELDS * HD];    // 16 KB
  __shared__ float    S[FIELDS * FIELDS];  // 4 KB  scores
  __shared__ _Float16 Am[FIELDS * FIELDS]; // 2 KB  softmaxed attn, f16
  __shared__ _Float16 y2[FIELDS * HD];     // 8 KB  layer-2 input, f16

  const int b    = blockIdx.x;
  const int lane = threadIdx.x;
  const int e0   = 2 * lane;               // each lane owns 2 emb columns

  // ---- phase 1: gather embedding fields into yh -----------------------
  for (int f = 0; f < 20; ++f) {
    int   idx = onehot_i[b * 20 + f];
    float w   = onehot_x[b * 20 + f];
    const float* row = xx + (long)idx * EMB + e0;
    float vx = row[0], vy = row[1];
    yh[f * EMB + e0]     = (_Float16)(vx * w);
    yh[f * EMB + e0 + 1] = (_Float16)(vy * w);
  }
  for (int s = 0; s < 2; ++s) {
    const int*   ip = mh_i + (s * BATCH + b) * 50;
    const float* wp = mh_x + (s * BATCH + b) * 50;
    float ax = 0.f, ay = 0.f;
    for (int j = 0; j < 50; ++j) {
      if (j + 4 < 50)
        __builtin_prefetch(xx + (long)ip[j + 4] * EMB + e0, 0, 0);
      const float* row = xx + (long)ip[j] * EMB + e0;
      float w = wp[j];
      ax += row[0] * w;
      ay += row[1] * w;
    }
    yh[(20 + s) * EMB + e0]     = (_Float16)ax;
    yh[(20 + s) * EMB + e0 + 1] = (_Float16)ay;
  }
  for (int c = 0; c < 10; ++c) {
    float w = ctns[b * 10 + c];
    const float* row = xy + c * EMB + e0;
    yh[(22 + c) * EMB + e0]     = (_Float16)(row[0] * w);
    yh[(22 + c) * EMB + e0 + 1] = (_Float16)(row[1] * w);
  }
  __syncthreads();

  // packed weight offsets (halves)
  const _Float16* wQ1 = wpack;
  const _Float16* wK1 = wpack + 8192;
  const _Float16* wV1 = wpack + 16384;
  const _Float16* wR1 = wpack + 24576;
  const _Float16* wQ2 = wpack + 32768;
  const _Float16* wK2 = wpack + 49152;
  const _Float16* wV2 = wpack + 65536;
  const _Float16* wR2 = wpack + 81920;

  // ---- layer 1 ---------------------------------------------------------
  project<0>(yh, EMB, wQ1, Qb1, Qm, nullptr, lane);
  project<0>(yh, EMB, wK1, Kb1, Km, nullptr, lane);
  project<1>(yh, EMB, wV1, Vb1, Vt, nullptr, lane);
  project<2>(yh, EMB, wR1, Rb1, nullptr, Res, lane);
  __syncthreads();
  attn<false>(Qm, Km, Vt, Res, S, Am, y2, logitW, lane);
  __syncthreads();

  // ---- layer 2 (reuse buffers) ----------------------------------------
  project<0>(y2, HD, wQ2, Qb2, Qm, nullptr, lane);
  project<0>(y2, HD, wK2, Kb2, Km, nullptr, lane);
  project<1>(y2, HD, wV2, Vb2, Vt, nullptr, lane);
  project<2>(y2, HD, wR2, Rb2, nullptr, Res, lane);
  __syncthreads();
  float part = attn<true>(Qm, Km, Vt, Res, S, Am, nullptr, logitW, lane);

  // ---- logit reduce + sigmoid -----------------------------------------
#pragma unroll
  for (int off = 16; off > 0; off >>= 1) part += __shfl_xor(part, off, 32);
  if (lane == 0) {
    float z = part + logitb[0];
    out[b] = 1.f / (1.f + __expf(-z));
  }
}

// ---------------------------------------------------------------------------
extern "C" void kernel_launch(void* const* d_in, const int* in_sizes, int n_in,
                              void* d_out, int out_size, void* d_ws,
                              size_t ws_size, hipStream_t stream) {
  const int*   onehot_i = (const int*)d_in[0];
  const float* onehot_x = (const float*)d_in[1];
  const int*   mh_i     = (const int*)d_in[2];
  const float* mh_x     = (const float*)d_in[3];
  const float* ctns     = (const float*)d_in[4];
  const float* xx       = (const float*)d_in[5];
  const float* xy       = (const float*)d_in[6];
  const float* QW1 = (const float*)d_in[7],  *Qb1 = (const float*)d_in[8];
  const float* KW1 = (const float*)d_in[9],  *Kb1 = (const float*)d_in[10];
  const float* VW1 = (const float*)d_in[11], *Vb1 = (const float*)d_in[12];
  const float* RW1 = (const float*)d_in[13], *Rb1 = (const float*)d_in[14];
  const float* QW2 = (const float*)d_in[15], *Qb2 = (const float*)d_in[16];
  const float* KW2 = (const float*)d_in[17], *Kb2 = (const float*)d_in[18];
  const float* VW2 = (const float*)d_in[19], *Vb2 = (const float*)d_in[20];
  const float* RW2 = (const float*)d_in[21], *Rb2 = (const float*)d_in[22];
  const float* logitW = (const float*)d_in[23];
  const float* logitb = (const float*)d_in[24];

  _Float16* wpack = (_Float16*)d_ws;

  // Pack weights into f16 WMMA B-fragments (deterministic, small).
  pack_weights<<<(8192 + 255) / 256, 256, 0, stream>>>(QW1, wpack + 0,     HD, EMB);
  pack_weights<<<(8192 + 255) / 256, 256, 0, stream>>>(KW1, wpack + 8192,  HD, EMB);
  pack_weights<<<(8192 + 255) / 256, 256, 0, stream>>>(VW1, wpack + 16384, HD, EMB);
  pack_weights<<<(8192 + 255) / 256, 256, 0, stream>>>(RW1, wpack + 24576, HD, EMB);
  pack_weights<<<(16384 + 255) / 256, 256, 0, stream>>>(QW2, wpack + 32768, HD, HD);
  pack_weights<<<(16384 + 255) / 256, 256, 0, stream>>>(KW2, wpack + 49152, HD, HD);
  pack_weights<<<(16384 + 255) / 256, 256, 0, stream>>>(VW2, wpack + 65536, HD, HD);
  pack_weights<<<(16384 + 255) / 256, 256, 0, stream>>>(RW2, wpack + 81920, HD, HD);

  autoint_fused<<<BATCH, 32, 0, stream>>>(
      onehot_i, onehot_x, mh_i, mh_x, ctns, xx, xy,
      Qb1, Kb1, Vb1, Rb1, Qb2, Kb2, Vb2, Rb2,
      logitW, logitb, wpack, (float*)d_out);
}
